// MSAColumnGlobalAttention_26242250179309
// MI455X (gfx1250) — matrix operand order repaired and checked by reading, not compiled, not run on hardware
//
#include <hip/hip_runtime.h>
#include <hip/hip_bf16.h>

// ---------------------------------------------------------------------------
// MSA column global attention, fused per-residue kernel for gfx1250 (CDNA5).
// bf16 WMMA 16x16x32 for KV / gate / output GEMMs, f32 accumulate.
// Weight B-fragments streamed through LDS double buffers via the Tensor Data
// Mover (tensor_load_to_lds + s_wait_tensorcnt), overlapping DMA with WMMA.
// ---------------------------------------------------------------------------

typedef __attribute__((ext_vector_type(16))) __bf16 v16bf;
typedef __attribute__((ext_vector_type(8)))  __bf16 v8bf;
typedef __attribute__((ext_vector_type(8)))  float  v8f;
typedef __attribute__((ext_vector_type(4)))  unsigned int v4u;
typedef __attribute__((ext_vector_type(8)))  int v8i;
typedef __attribute__((ext_vector_type(4)))  int v4i;

#define S_DIM 1024
#define R_DIM 512
#define C_DIM 256
#define H_DIM 8
#define CH_DIM 32
#define HC_DIM 256

// LDS layout (bytes). Padded strides to avoid 64-bank conflicts.
#define KVP   68                     // kv row stride in halves (32 k + 32 v + pad)
#define STGP  272                    // stage row stride in halves (256 + 16 pad)
#define OFF_KV     0
#define SZ_KV      (S_DIM * KVP * 2)                 // 139264
#define OFF_STAGE  (OFF_KV + SZ_KV)
#define SZ_STAGE   (8 * 16 * STGP * 2)               // 69632
#define OFF_WL     (OFF_STAGE + SZ_STAGE)            // softmax wl; reused as B dbl-buf
#define SZ_WL      (H_DIM * S_DIM * 4)               // 32768 = 2 x 16KB buffers
#define OFF_KVB    (OFF_WL + SZ_WL)                  // KV weight fragments (phase 1)
#define SZ_KVB     32768
#define OFF_STATS  (OFF_KVB + SZ_KVB)
#define SZ_STATS   (S_DIM * 2 * 4)                   // 8192
#define OFF_MROW   (OFF_STATS + SZ_STATS)
#define SZ_MROW    (S_DIM * 4)                       // 4096
#define OFF_QAVG   (OFF_MROW + SZ_MROW)
#define OFF_QH     (OFF_QAVG + 1024)
#define OFF_OLDS   (OFF_QH + 1024)
#define OFF_MSUM   (OFF_OLDS + 1024)
#define LDS_BYTES  (OFF_MSUM + 16)                   // ~283 KB (< 320 KB WGP max)

#define BUF0_OFF   (OFF_WL)                          // phase-4 B double buffer LDS
#define BUF1_OFF   (OFF_WL + 16384)                  //   byte addresses (dyn-LDS base 0)

static __device__ __forceinline__ __bf16 f2bf(float f) {
  unsigned u = __builtin_bit_cast(unsigned, f);
  u += 0x7FFFu + ((u >> 16) & 1u);                  // round-to-nearest-even
  unsigned short h = (unsigned short)(u >> 16);
  return __builtin_bit_cast(__bf16, h);
}
static __device__ __forceinline__ float bf2f(__bf16 b) {
  unsigned short h = __builtin_bit_cast(unsigned short, b);
  unsigned u = ((unsigned)h) << 16;
  return __builtin_bit_cast(float, u);
}
static __device__ __forceinline__ float wred_sum(float v) {
#pragma unroll
  for (int o = 16; o > 0; o >>= 1) v += __shfl_xor(v, o);
  return v;
}
static __device__ __forceinline__ float wred_max(float v) {
#pragma unroll
  for (int o = 16; o > 0; o >>= 1) v = fmaxf(v, __shfl_xor(v, o));
  return v;
}

// ---------------------------------------------------------------------------
// TDM: DMA a contiguous block of n8 8-byte elements from global to LDS.
// D# per CDNA5 ISA ch.8: group0 = {count=1 | lds_addr | global_addr | type=2},
// group1 = {data_size=8B, tensor_dim0 = tile_dim0 = n8, tile_dim1 = 1,
//           tensor_dim0_stride = n8}; groups 2/3 zero (2D descriptor).
// Issued wave-uniform (EXEC ignored); tracked with TENSORcnt.
// This toolchain's builtin takes 6 args: (g0, g1, g2, g3, g_extra, cpol).
// ---------------------------------------------------------------------------
static __device__ __forceinline__ void tdm_load_block(unsigned lds_off,
                                                      const void* gaddr,
                                                      unsigned n8) {
  unsigned long long ga = (unsigned long long)(uintptr_t)gaddr;
  v4u g0;
  g0[0] = 1u;                                        // count=1, user mode, no gather
  g0[1] = lds_off;                                   // LDS byte address
  g0[2] = (unsigned)(ga & 0xFFFFFFFFu);              // global_addr[31:0]
  g0[3] = (unsigned)((ga >> 32) & 0x01FFFFFFu)       // global_addr[56:32]
          | (2u << 30);                              // type = 2 ("image")
  v8i g1;
  g1[0] = (3 << 16);                                 // data_size = 8B; mask/flags 0
  g1[1] = (int)(n8 << 16);                           // tensor_dim0[15:0] (bits 63:48)
  g1[2] = (1 << 16);                                 // tensor_dim0 hi=0 ; tensor_dim1=1
  g1[3] = (int)(n8 << 16);                           // tensor_dim1 hi=0 ; tile_dim0=n8
  g1[4] = 1;                                         // tile_dim1 = 1, tile_dim2 = 0
  g1[5] = (int)n8;                                   // tensor_dim0_stride[31:0]
  g1[6] = 0;                                         // stride0 hi / stride1 lo
  g1[7] = 0;                                         // stride1 hi
  v4i z4 = {0, 0, 0, 0};
  v8i z8 = {0, 0, 0, 0, 0, 0, 0, 0};
  __builtin_amdgcn_tensor_load_to_lds(g0, g1, z4, z4, z8, 0);
}

// A-fragment (16x32 bf16): lane L holds row M=L%16,
// halves 0..15 -> K = (L/16)*16 + i (matches B-fragment order).
static __device__ __forceinline__ v16bf load_a_frag(const __bf16* stg, int lane, int kt) {
  const int row = lane & 15;
  const int kbase = kt * 32 + ((lane >> 4) << 4);
  const __bf16* p = stg + row * STGP + kbase;
  v16bf a = *(const v16bf*)p;
  return a;
}

static __device__ __forceinline__ v8f wmma_bf16(v16bf a, v16bf b, v8f c) {
  return __builtin_amdgcn_wmma_f32_16x16x32_bf16(false, a, false, b, (short)0, c,
                                                 false, false);
}

// ---------------------------------------------------------------------------
// Prep: swizzle Wk|Wv, Wg, Wf (f32) into bf16 B-fragment layout.
// B-fragment (32x16): lane L -> N=L%16, half i -> K = (L/16)*16 + i.
// ws element index: ((kt*NT + nt)*32 + L)*16 + i
// ---------------------------------------------------------------------------
__global__ void msa_prep_weights(const float* __restrict__ Wk, const float* __restrict__ Wv,
                                 const float* __restrict__ Wg, const float* __restrict__ Wf,
                                 __bf16* __restrict__ wsKV, __bf16* __restrict__ wsWg,
                                 __bf16* __restrict__ wsWf) {
  int gid = blockIdx.x * 256 + threadIdx.x;
  if (gid < 16384) {                                   // KV: 8 kt x 4 nt
    int i = gid & 15, L = (gid >> 4) & 31, nt = (gid >> 9) & 3, kt = gid >> 11;
    int K = kt * 32 + ((L >> 4) << 4) + i;
    int N = nt * 16 + (L & 15);
    float v = (N < 32) ? Wk[K * 32 + N] : Wv[K * 32 + (N - 32)];
    wsKV[gid] = f2bf(v);
  } else if (gid < 16384 + 65536) {                    // Wg: 8 kt x 16 nt
    int g = gid - 16384;
    int i = g & 15, L = (g >> 4) & 31, nt = (g >> 9) & 15, kt = g >> 13;
    int K = kt * 32 + ((L >> 4) << 4) + i;
    int N = nt * 16 + (L & 15);
    wsWg[g] = f2bf(Wg[K * HC_DIM + N]);
  } else if (gid < 16384 + 2 * 65536) {                // Wf: 8 kt x 16 nt
    int g = gid - (16384 + 65536);
    int i = g & 15, L = (g >> 4) & 31, nt = (g >> 9) & 15, kt = g >> 13;
    int K = kt * 32 + ((L >> 4) << 4) + i;
    int N = nt * 16 + (L & 15);
    wsWf[g] = f2bf(Wf[K * C_DIM + N]);
  }
}

// ---------------------------------------------------------------------------
// Main fused kernel: one block per residue r. 256 threads = 8 waves.
// ---------------------------------------------------------------------------
__global__ __launch_bounds__(256, 1)
void msa_colattn_main(const float* __restrict__ x,      // [S,R,C]
                      const float* __restrict__ mask,   // [S,R]
                      const float* __restrict__ ls, const float* __restrict__ lb,
                      const float* __restrict__ Wq,     // [C, HC]
                      const float* __restrict__ bg,     // [HC]
                      const float* __restrict__ bfv,    // [C]
                      const __bf16* __restrict__ wsKV,
                      const __bf16* __restrict__ wsWg,
                      const __bf16* __restrict__ wsWf,
                      float* __restrict__ out) {        // [S,R,C]
  extern __shared__ char smem[];
  __bf16* kv    = (__bf16*)(smem + OFF_KV);
  __bf16* stage = (__bf16*)(smem + OFF_STAGE);
  float*  wl    = (float*)(smem + OFF_WL);
  __bf16* bbuf0 = (__bf16*)(smem + OFF_WL);            // phase-4 alias of wl
  __bf16* bbuf1 = bbuf0 + 8192;
  __bf16* kvb   = (__bf16*)(smem + OFF_KVB);
  float*  stats = (float*)(smem + OFF_STATS);
  float*  mrow  = (float*)(smem + OFF_MROW);
  float*  qavg  = (float*)(smem + OFF_QAVG);
  float*  qh    = (float*)(smem + OFF_QH);
  float*  olds  = (float*)(smem + OFF_OLDS);
  float*  msum  = (float*)(smem + OFF_MSUM);

  const int r    = blockIdx.x;
  const int tid  = threadIdx.x;
  const int lane = tid & 31;
  const int wid  = tid >> 5;
  __bf16* stg = stage + wid * (16 * STGP);

  // ---- phase 0: zero accumulators, DMA KV weight frags, cache mask ---------
  if (wid == 0) tdm_load_block(OFF_KVB, wsKV, SZ_KVB / 8);   // 32KB via TDM
  qavg[tid] = 0.f;
  if (tid == 0) *msum = 0.f;
  __syncthreads();
  {
    float lm = 0.f;
    for (int s = tid; s < S_DIM; s += 256) {
      float m = mask[(size_t)s * R_DIM + r];
      mrow[s] = m;
      lm += m;
    }
    atomicAdd(msum, lm);
  }
  // per-lane LayerNorm affine params for c = lane*8 .. lane*8+7
  float gs[8], gb[8];
#pragma unroll
  for (int j = 0; j < 8; ++j) { gs[j] = ls[lane * 8 + j]; gb[j] = lb[lane * 8 + j]; }
  if (wid == 0) __builtin_amdgcn_s_wait_tensorcnt(0);
  __syncthreads();

  // ---- phase 1: LayerNorm + KV GEMM per 16-row s-tile ----------------------
  float qacc[8];
#pragma unroll
  for (int j = 0; j < 8; ++j) qacc[j] = 0.f;

  for (int st = wid; st < S_DIM / 16; st += 8) {
    for (int m = 0; m < 16; ++m) {
      const int s = st * 16 + m;
      const float* xr = x + ((size_t)s * R_DIM + r) * C_DIM;
      if (m < 15) __builtin_prefetch(xr + (size_t)R_DIM * C_DIM, 0, 0);
      float4 a0 = *(const float4*)(xr + lane * 8);
      float4 a1 = *(const float4*)(xr + lane * 8 + 4);
      float xs[8] = {a0.x, a0.y, a0.z, a0.w, a1.x, a1.y, a1.z, a1.w};
      float sum = 0.f, sq = 0.f;
#pragma unroll
      for (int j = 0; j < 8; ++j) { sum += xs[j]; sq += xs[j] * xs[j]; }
      sum = wred_sum(sum);
      sq  = wred_sum(sq);
      const float mean = sum * (1.f / 256.f);
      const float var  = sq * (1.f / 256.f) - mean * mean;
      const float rstd = rsqrtf(var + 1e-5f);
      if (lane == 0) { stats[2 * s] = mean; stats[2 * s + 1] = rstd; }
      const float mv = mrow[s];
      v8bf pk;
#pragma unroll
      for (int j = 0; j < 8; ++j) {
        float xn = (xs[j] - mean) * rstd * gs[j] + gb[j];
        qacc[j] += xn * mv;
        pk[j] = f2bf(xn);
      }
      *(v8bf*)(stg + m * STGP + lane * 8) = pk;
    }
    // KV GEMM: [16 x 256] x [256 x 64], B from LDS
    v8f acc[4];
#pragma unroll
    for (int nt = 0; nt < 4; ++nt)
#pragma unroll
      for (int j = 0; j < 8; ++j) acc[nt][j] = 0.f;
    for (int kt = 0; kt < 8; ++kt) {
      v16bf a = load_a_frag(stg, lane, kt);
#pragma unroll
      for (int nt = 0; nt < 4; ++nt) {
        v16bf b = *((const v16bf*)kvb + ((kt * 4 + nt) * 32 + lane));
        acc[nt] = wmma_bf16(a, b, acc[nt]);
      }
    }
#pragma unroll
    for (int nt = 0; nt < 4; ++nt)
#pragma unroll
      for (int j = 0; j < 8; ++j) {
        int srow = st * 16 + j + ((lane >> 4) << 3);
        int col  = nt * 16 + (lane & 15);
        kv[srow * KVP + col] = f2bf(acc[nt][j]);
      }
  }
#pragma unroll
  for (int j = 0; j < 8; ++j) atomicAdd(&qavg[lane * 8 + j], qacc[j]);
  __syncthreads();

  // ---- global query: q = (q_avg @ Wq) * CH^-0.5 ----------------------------
  {
    float inv = 1.f / (*msum + 1e-10f);
    qavg[tid] *= inv;
  }
  __syncthreads();
  {
    float a = 0.f;
    for (int c = 0; c < C_DIM; ++c) a += qavg[c] * Wq[c * HC_DIM + tid];
    qh[tid] = a * 0.17677669529663687f;   // 32^-0.5
  }
  __syncthreads();

  // ---- phase 2+3: attention, one wave per head -----------------------------
  {
    const int h = wid;
    float* wlh = wl + h * S_DIM;
    float mx = -3.4e38f;
    for (int si = lane; si < S_DIM; si += 32) {
      float dot = 0.f;
#pragma unroll 8
      for (int ch = 0; ch < CH_DIM; ++ch)
        dot += qh[h * CH_DIM + ch] * bf2f(kv[si * KVP + ch]);
      float lg = dot + 1e9f * (mrow[si] - 1.f);
      wlh[si] = lg;
      mx = fmaxf(mx, lg);
    }
    mx = wred_max(mx);
    float sum = 0.f;
    for (int si = lane; si < S_DIM; si += 32) {
      float e = __expf(wlh[si] - mx);
      wlh[si] = e;
      sum += e;
    }
    sum = wred_sum(sum);
    const float winv = 1.f / sum;
    // o[h, ch=lane] = sum_s w * v
    float o = 0.f;
    for (int si = 0; si < S_DIM; ++si)
      o += wlh[si] * bf2f(kv[si * KVP + 32 + lane]);
    olds[h * CH_DIM + lane] = o * winv;
  }
  __syncthreads();   // wl region is dead past here; reused as B double buffer

  // ---- phase 4: gate GEMM + sigmoid*o + output GEMM ------------------------
  // 8 tile-groups; wave w owns tile st = g*8 + w. Wg/Wf B-fragments streamed
  // through a 2x16KB LDS double buffer by the Tensor Data Mover (wave 0
  // issues the DMA for kt+1 while all waves run WMMAs on kt).
  float bgl[16], bfl[16], olv[16];
#pragma unroll
  for (int nt = 0; nt < 16; ++nt) {
    int n = nt * 16 + (lane & 15);
    bgl[nt] = bg[n];
    bfl[nt] = bfv[n];
    olv[nt] = olds[n];
  }

  for (int g = 0; g < 8; ++g) {
    const int st = g * 8 + wid;
    if (wid == 0) tdm_load_block(BUF0_OFF, wsWg, 2048);      // gate kt=0
    // rebuild xn tile from cached stats (wave-private stage)
    for (int m = 0; m < 16; ++m) {
      const int s = st * 16 + m;
      const float* xr = x + ((size_t)s * R_DIM + r) * C_DIM;
      if (m < 15) __builtin_prefetch(xr + (size_t)R_DIM * C_DIM, 0, 0);
      float4 a0 = *(const float4*)(xr + lane * 8);
      float4 a1 = *(const float4*)(xr + lane * 8 + 4);
      float xs[8] = {a0.x, a0.y, a0.z, a0.w, a1.x, a1.y, a1.z, a1.w};
      const float mean = stats[2 * s], rstd = stats[2 * s + 1];
      v8bf pk;
#pragma unroll
      for (int j = 0; j < 8; ++j)
        pk[j] = f2bf((xs[j] - mean) * rstd * gs[j] + gb[j]);
      *(v8bf*)(stg + m * STGP + lane * 8) = pk;
    }
    // ---- gate GEMM: [16 x 256] x [256 x 256], TDM-streamed B ----
    if (wid == 0) __builtin_amdgcn_s_wait_tensorcnt(0);
    __syncthreads();
    v8f accg[16];
#pragma unroll
    for (int nt = 0; nt < 16; ++nt)
#pragma unroll
      for (int j = 0; j < 8; ++j) accg[nt][j] = 0.f;
    for (int kt = 0; kt < 8; ++kt) {
      if (kt < 7 && wid == 0)
        tdm_load_block(((kt + 1) & 1) ? BUF1_OFF : BUF0_OFF,
                       wsWg + (kt + 1) * 8192, 2048);
      const __bf16* bb = (kt & 1) ? bbuf1 : bbuf0;
      v16bf a = load_a_frag(stg, lane, kt);
#pragma unroll
      for (int nt = 0; nt < 16; ++nt) {
        v16bf b = *((const v16bf*)bb + (nt * 32 + lane));
        accg[nt] = wmma_bf16(a, b, accg[nt]);
      }
      if (wid == 0) __builtin_amdgcn_s_wait_tensorcnt(0);
      __syncthreads();
    }
    // prefetch Wf kt=0 (buf0 free: last used at kt==6, synced) while gating
    if (wid == 0) tdm_load_block(BUF0_OFF, wsWf, 2048);
    // G[s, n] = o[n] * sigmoid(gate + bg[n]); restage as bf16 (wave-private)
#pragma unroll
    for (int nt = 0; nt < 16; ++nt)
#pragma unroll
      for (int j = 0; j < 8; ++j) {
        float gv = accg[nt][j] + bgl[nt];
        float sg = 1.f / (1.f + __expf(-gv));
        int row = j + ((lane >> 4) << 3);
        stg[row * STGP + nt * 16 + (lane & 15)] = f2bf(olv[nt] * sg);
      }
    if (wid == 0) __builtin_amdgcn_s_wait_tensorcnt(0);
    __syncthreads();
    // ---- output GEMM: [16 x 256] x [256 x 256], TDM-streamed B ----
    v8f acco[16];
#pragma unroll
    for (int nt = 0; nt < 16; ++nt)
#pragma unroll
      for (int j = 0; j < 8; ++j) acco[nt][j] = 0.f;
    for (int kt = 0; kt < 8; ++kt) {
      if (kt < 7 && wid == 0)
        tdm_load_block(((kt + 1) & 1) ? BUF1_OFF : BUF0_OFF,
                       wsWf + (kt + 1) * 8192, 2048);
      const __bf16* bb = (kt & 1) ? bbuf1 : bbuf0;
      v16bf a = load_a_frag(stg, lane, kt);
#pragma unroll
      for (int nt = 0; nt < 16; ++nt) {
        v16bf b = *((const v16bf*)bb + (nt * 32 + lane));
        acco[nt] = wmma_bf16(a, b, acco[nt]);
      }
      if (wid == 0) __builtin_amdgcn_s_wait_tensorcnt(0);
      __syncthreads();
    }
#pragma unroll
    for (int nt = 0; nt < 16; ++nt)
#pragma unroll
      for (int j = 0; j < 8; ++j) {
        int srow = st * 16 + j + ((lane >> 4) << 3);
        int cc   = nt * 16 + (lane & 15);
        out[((size_t)srow * R_DIM + r) * C_DIM + cc] = acco[nt][j] + bfl[nt];
      }
  }
}

// ---------------------------------------------------------------------------
extern "C" void kernel_launch(void* const* d_in, const int* in_sizes, int n_in,
                              void* d_out, int out_size, void* d_ws, size_t ws_size,
                              hipStream_t stream) {
  const float* x    = (const float*)d_in[0];
  const float* mask = (const float*)d_in[1];
  const float* ls   = (const float*)d_in[2];
  const float* lb   = (const float*)d_in[3];
  const float* Wq   = (const float*)d_in[4];
  const float* Wk   = (const float*)d_in[5];
  const float* Wv   = (const float*)d_in[6];
  const float* Wg   = (const float*)d_in[7];
  const float* bg   = (const float*)d_in[8];
  const float* Wf   = (const float*)d_in[9];
  const float* bf   = (const float*)d_in[10];
  float* out = (float*)d_out;

  __bf16* wsKV = (__bf16*)d_ws;            // 16384 halves
  __bf16* wsWg = wsKV + 16384;             // 65536 halves
  __bf16* wsWf = wsWg + 65536;             // 65536 halves

  (void)hipFuncSetAttribute((const void*)msa_colattn_main,
                            hipFuncAttributeMaxDynamicSharedMemorySize, LDS_BYTES);

  msa_prep_weights<<<(147456 + 255) / 256, 256, 0, stream>>>(Wk, Wv, Wg, Wf,
                                                             wsKV, wsWg, wsWf);
  msa_colattn_main<<<R_DIM, 256, LDS_BYTES, stream>>>(x, mask, ls, lb, Wq, bg, bf,
                                                      wsKV, wsWg, wsWf, out);
}
